// AutoregressiveMultiGNNv1_50268297232897
// MI455X (gfx1250) — matrix operand order repaired and checked by reading, not compiled, not run on hardware
//
#include <hip/hip_runtime.h>

// ---------------------------------------------------------------------------
// Problem constants (from the reference)
// ---------------------------------------------------------------------------
#define NN     10000   // nodes
#define NC     3       // conformers
#define NE     100000  // edges
#define NS     128     // node scalar width
#define NV     16      // node vector channels
#define SE     32      // edge scalar width
#define VE     1       // edge vector channels
#define NHEADS 4
#define ODIM   4

typedef unsigned short u16;
typedef __attribute__((ext_vector_type(16))) __bf16 v16bf;
typedef __attribute__((ext_vector_type(8)))  float  v8f;

__device__ __forceinline__ u16 f2bf(float f) {
  union { float f; unsigned u; } x; x.f = f;
  unsigned r = x.u + 0x7FFFu + ((x.u >> 16) & 1u);   // round-to-nearest-even
  return (u16)(r >> 16);
}

// ---------------------------------------------------------------------------
// WMMA GEMM: C[M,N] = act(A[M,kp](bf16) * Bp + bias + Res)
//   A  : row-major, leading dim kp = roundup(K,32); pad columns are zero.
//   Bp : weights pre-packed [n][k], n < roundup(N,64), k < kp, zero padded.
//   act: 0 = none, 1 = SiLU, 2 = sigmoid.  Writes f32 (Cf, ld=N) and/or
//   bf16 (Cb, ld=ldcb with zeroed pad columns).
// Block = 256 threads = 8 waves (2x4), tile 32x64, K chunk 32.
// A tile is double-buffered in LDS and staged with the CDNA5 async
// global->LDS DMA (ASYNCcnt), so the copy of tile i+1 overlaps the
// ds_load fragments + v_wmma of tile i; one barrier per K step.
// ---------------------------------------------------------------------------
#define TM 32
#define TN 64
#define KC 32
#define LLD 40   // padded LDS row stride (ushorts): 80B, 16B-aligned fragments

__global__ __launch_bounds__(256)
void k_gemm(const u16* __restrict__ A, const u16* __restrict__ B,
            const float* __restrict__ bias,
            float* __restrict__ Cf, u16* __restrict__ Cb, int ldcb,
            int M, int N, int kp, int act, const float* __restrict__ Res)
{
  __shared__ __align__(16) u16 As[2][TM * LLD];
  const int tid  = threadIdx.x;
  const int lane = tid & 31;
  const int wave = tid >> 5;
  const int wm = wave >> 2;      // 0..1
  const int wn = wave & 3;       // 0..3
  const int m0 = blockIdx.y * TM;
  const int n0 = blockIdx.x * TN;

  // A staging: one b64 per thread per K chunk, row clamped (epilogue masks)
  const int r  = tid >> 3;
  const int kc = (tid & 7) << 2;
  long gm = m0 + r; if (gm >= M) gm = M - 1;
  const u16* aptr = A + gm * (size_t)kp + kc;
  // LDS byte offsets for the async DMA destination (generic ptr low 32 bits
  // are the LDS offset inside the shared aperture)
  const unsigned lds0 = (unsigned)(size_t)(void*)&As[0][r * LLD + kc];
  const unsigned lds1 = (unsigned)(size_t)(void*)&As[1][r * LLD + kc];
  // fragment addresses
  const int arow  = wm * 16 + (lane & 15);
  const int khalf = (lane >> 4) << 3;
  const u16* bptr = B + (size_t)(n0 + wn * 16 + (lane & 15)) * kp
                      + ((lane >> 4) << 3);

  // prologue: async-copy tile 0 into buffer 0
  asm volatile("global_load_async_to_lds_b64 %0, %1, off"
               :: "v"(lds0), "v"(aptr) : "memory");

  v8f acc = {0.f, 0.f, 0.f, 0.f, 0.f, 0.f, 0.f, 0.f};
  const int nt = kp / KC;
  for (int i = 0; i < nt; ++i) {
    const int k0 = i * KC;
    // my tile-i copy done; barrier -> everyone's tile-i copy visible
    asm volatile("s_wait_asynccnt 0x0" ::: "memory");
    __syncthreads();
    // kick off tile i+1 into the other buffer (its last readers finished
    // before the barrier above), overlapping with this tile's math
    if (i + 1 < nt) {
      asm volatile("global_load_async_to_lds_b64 %0, %1, off"
                   :: "v"((i & 1) ? lds0 : lds1), "v"(aptr + k0 + KC)
                   : "memory");
    }
    union { uint4 q[2]; v16bf v; } ua, ub;
    if (i & 1) {
      ua.q[0] = *(const uint4*)&As[1][arow * LLD + khalf];
      ua.q[1] = *(const uint4*)&As[1][arow * LLD + khalf + 16];
    } else {
      ua.q[0] = *(const uint4*)&As[0][arow * LLD + khalf];
      ua.q[1] = *(const uint4*)&As[0][arow * LLD + khalf + 16];
    }
    ub.q[0] = *(const uint4*)(bptr + k0);
    ub.q[1] = *(const uint4*)(bptr + k0 + 16);
    acc = __builtin_amdgcn_wmma_f32_16x16x32_bf16(false, ua.v, false, ub.v,
                                                  (short)0, acc, false, false);
  }

  // C/D layout: VGPR i, lanes 0-15 -> M=i, lanes 16-31 -> M=i+8; N = lane&15
  const int ncol  = n0 + wn * 16 + (lane & 15);
  const int mbase = m0 + wm * 16 + ((lane >> 4) << 3);
  #pragma unroll
  for (int i = 0; i < 8; ++i) {
    int m = mbase + i;
    if (m < M) {
      if (ncol < N) {
        float x = acc[i];
        if (bias) x += bias[ncol];
        if (Res)  x += Res[(long)m * N + ncol];
        if (act == 1)      x = x / (1.f + __expf(-x));
        else if (act == 2) x = 1.f / (1.f + __expf(-x));
        if (Cf) Cf[(long)m * N + ncol] = x;
        if (Cb) Cb[(long)m * ldcb + ncol] = f2bf(x);
      } else if (Cb && ncol < ldcb) {
        Cb[(long)m * ldcb + ncol] = 0;   // keep bf16 pad columns zero
      }
    }
  }
}

// ---------------------------------------------------------------------------
// Elementwise / reduction kernels
// ---------------------------------------------------------------------------
__global__ void k_cvt(const float* __restrict__ x, u16* __restrict__ y, long n) {
  long i = (long)blockIdx.x * blockDim.x + threadIdx.x;
  if (i < n) y[i] = f2bf(x[i]);
}

// pack f32 weight (K,N) row-major -> bf16 [npad][kp], zero padded
__global__ void k_pack(const float* __restrict__ w, u16* __restrict__ out,
                       int K, int N, int kp, long tot)
{
  long i = (long)blockIdx.x * blockDim.x + threadIdx.x;
  if (i >= tot) return;
  long n = i / kp; int k = (int)(i % kp);
  out[i] = (n < N && k < K) ? f2bf(w[(long)k * N + n]) : (u16)0;
}

// zero pad columns [w, ld) of a bf16 matrix
__global__ void k_padz(u16* __restrict__ m, long rows, int w, int ld)
{
  int pw = ld - w;
  long i = (long)blockIdx.x * blockDim.x + threadIdx.x;
  if (i >= rows * pw) return;
  long r = i / pw; int c = (int)(i % pw);
  m[r * ld + w + c] = 0;
}

// LayerNorm over last dim d; one wave per row (wave32 shuffles)
__global__ void k_ln(const float* __restrict__ x, const float* __restrict__ gg,
                     const float* __restrict__ bb, float* __restrict__ yf,
                     u16* __restrict__ yb, long rows, int d)
{
  long row = (long)blockIdx.x * (blockDim.x >> 5) + (threadIdx.x >> 5);
  int lane = threadIdx.x & 31;
  if (row >= rows) return;
  const float* xr = x + row * d;
  float s1 = 0.f, s2 = 0.f;
  for (int i = lane; i < d; i += 32) { float t = xr[i]; s1 += t; s2 += t * t; }
  for (int off = 16; off > 0; off >>= 1) {
    s1 += __shfl_down(s1, off, 32); s2 += __shfl_down(s2, off, 32);
  }
  s1 = __shfl(s1, 0, 32); s2 = __shfl(s2, 0, 32);
  float mu  = s1 / d;
  float inv = rsqrtf(fmaxf(s2 / d - mu * mu, 0.f) + 1e-5f);
  for (int i = lane; i < d; i += 32) {
    float t = (xr[i] - mu) * inv * gg[i] + bb[i];
    if (yf) yf[row * d + i] = t;
    if (yb) yb[row * d + i] = f2bf(t);
  }
}

// GVP-norm vector part: v / sqrt(mean_c max(||v_c||^2, eps))
__global__ void k_vnormv(const float* __restrict__ v, float* __restrict__ out,
                         long rows, int nv)
{
  long r = (long)blockIdx.x * blockDim.x + threadIdx.x;
  if (r >= rows) return;
  const float* vr = v + r * nv * 3;
  float acc = 0.f;
  for (int c = 0; c < nv; ++c) {
    float a = vr[3 * c], b = vr[3 * c + 1], d = vr[3 * c + 2];
    acc += fmaxf(a * a + b * b + d * d, 1e-8f);
  }
  float inv = rsqrtf(acc / nv);
  float* o = out + r * nv * 3;
  for (int i = 0; i < nv * 3; ++i) o[i] = vr[i] * inv;
}

// (rows, vi, 3) f32 -> (rows*3, ld) bf16 with zero pad (swapaxes for v^T @ wh)
__global__ void k_vT(const float* __restrict__ v, u16* __restrict__ vt,
                     long rows, int vi, int ld)
{
  long i = (long)blockIdx.x * blockDim.x + threadIdx.x;
  long tot = rows * 3 * ld;
  if (i >= tot) return;
  long r3 = i / ld; int c = (int)(i % ld);
  long row = r3 / 3; int comp = (int)(r3 % 3);
  vt[i] = (c < vi) ? f2bf(v[(row * vi + c) * 3 + comp]) : (u16)0;
}

// copy bf16 column block into wider bf16 matrix
__global__ void k_copyc(const u16* __restrict__ s, int sld, u16* __restrict__ d,
                        int dld, long rows, int cols, int doff)
{
  long i = (long)blockIdx.x * blockDim.x + threadIdx.x;
  if (i >= rows * cols) return;
  long r = i / cols; int c = (int)(i % cols);
  d[r * dld + doff + c] = s[r * sld + c];
}

// ||vh||-over-3 -> bf16 columns of concat matrix
__global__ void k_vhnorm(const float* __restrict__ vh, u16* __restrict__ d,
                         int dld, int doff, long rows, int h)
{
  long i = (long)blockIdx.x * blockDim.x + threadIdx.x;
  if (i >= rows * h) return;
  long r = i / h; int c = (int)(i % h);
  float a = vh[(r * 3 + 0) * h + c];
  float b = vh[(r * 3 + 1) * h + c];
  float e = vh[(r * 3 + 2) * h + c];
  d[r * dld + doff + c] = f2bf(sqrtf(fmaxf(a * a + b * b + e * e, 1e-8f)));
}

// y = act(x) (+res), f32/bf16 outputs (width is a multiple of 32 at all uses)
__global__ void k_act(const float* __restrict__ x, const float* __restrict__ res,
                      float* __restrict__ yf, u16* __restrict__ yb, long n, int act)
{
  long i = (long)blockIdx.x * blockDim.x + threadIdx.x;
  if (i >= n) return;
  float v = x[i];
  if (act == 1) v = v / (1.f + __expf(-v));
  if (res) v += res[i];
  if (yf) yf[i] = v;
  if (yb) yb[i] = f2bf(v);
}

// (rows*3, vo) -> (rows, vo, 3), optional sigmoid gate and residual
__global__ void k_gatev(const float* __restrict__ vot, const float* __restrict__ gate,
                        const float* __restrict__ res, float* __restrict__ out,
                        long rows, int vo)
{
  long i = (long)blockIdx.x * blockDim.x + threadIdx.x;
  long tot = rows * vo * 3;
  if (i >= tot) return;
  long r = i / (vo * 3); int rem = (int)(i % (vo * 3));
  int c = rem / 3, d = rem % 3;
  float x = vot[(r * 3 + d) * vo + c];
  if (gate) x *= gate[r * vo + c];
  if (res)  x += res[i];
  out[i] = x;
}

// edge-message scalar concat [S[src], es, S[dst]] with decoder fwd-select
__global__ void k_msgs(const u16* __restrict__ sn, const u16* __restrict__ an,
                       const u16* __restrict__ es, const int* __restrict__ src,
                       const int* __restrict__ dst, u16* __restrict__ out,
                       long E, int C, int ns, int se, int es_ld, int out_ld)
{
  int W = 2 * ns + se;
  long i = (long)blockIdx.x * blockDim.x + threadIdx.x;
  if (i >= E * C * out_ld) return;
  long r = i / out_ld; int c = (int)(i % out_ld);
  if (c >= W) { out[i] = 0; return; }
  long e = r / C; int cf = (int)(r % C);
  bool fwd = src[e] < dst[e];
  u16 val;
  if (c < ns) {
    const u16* S = (fwd || !an) ? sn : an;
    val = S[((long)src[e] * C + cf) * ns + c];
  } else if (c < ns + se) {
    val = es[r * es_ld + (c - ns)];
  } else {
    const u16* S = (fwd || !an) ? sn : an;
    val = S[((long)dst[e] * C + cf) * ns + (c - ns - se)];
  }
  out[i] = val;
}

// edge-message vector concat [V[src], ev, V[dst]] (f32)
__global__ void k_msgv(const float* __restrict__ vn, const float* __restrict__ av,
                       const float* __restrict__ ev, const int* __restrict__ src,
                       const int* __restrict__ dst, float* __restrict__ out,
                       long E, int C, int nv, int ve)
{
  int W = (2 * nv + ve) * 3;
  long i = (long)blockIdx.x * blockDim.x + threadIdx.x;
  if (i >= E * C * W) return;
  long r = i / W; int c = (int)(i % W);
  long e = r / C; int cf = (int)(r % C);
  int ch = c / 3, d = c % 3;
  bool fwd = src[e] < dst[e];
  float val;
  if (ch < nv) {
    const float* S = (fwd || !av) ? vn : av;
    val = S[(((long)src[e] * C + cf) * nv + ch) * 3 + d];
  } else if (ch < nv + ve) {
    val = ev[(r * ve + (ch - nv)) * 3 + d];
  } else {
    const float* S = (fwd || !av) ? vn : av;
    val = S[(((long)dst[e] * C + cf) * nv + (ch - nv - ve)) * 3 + d];
  }
  out[i] = val;
}

__global__ void k_scat(const float* __restrict__ m, const int* __restrict__ dst,
                       float* __restrict__ out, long E, int C, int w)
{
  long i = (long)blockIdx.x * blockDim.x + threadIdx.x;
  if (i >= E * C * w) return;
  long r = i / w; int cc = (int)(i % w);
  long e = r / C; int cf = (int)(r % C);
  atomicAdd(&out[((long)dst[e] * C + cf) * w + cc], m[i]);
}

__global__ void k_count(const int* __restrict__ dst, float* __restrict__ cnt, long E)
{
  long e = (long)blockIdx.x * blockDim.x + threadIdx.x;
  if (e < E) atomicAdd(&cnt[dst[e]], 1.f);
}

__global__ void k_addsc(const float* __restrict__ agg, const float* __restrict__ cnt,
                        float* __restrict__ s, long rows, int C, int w)
{
  long i = (long)blockIdx.x * blockDim.x + threadIdx.x;
  if (i >= rows * w) return;
  long r = i / w; int node = (int)(r / C);
  s[i] += agg[i] / fmaxf(cnt[node], 1.f);
}

// softmax attention over the C(=3) conformers; one wave per (node, head)
__global__ void k_attn(const float* __restrict__ q, const float* __restrict__ k,
                       const float* __restrict__ v, u16* __restrict__ outb,
                       long N, int C, int d, int heads)
{
  long gt = (long)blockIdx.x * blockDim.x + threadIdx.x;
  long wid = gt >> 5;
  int lane = threadIdx.x & 31;
  if (wid >= N * heads) return;
  long node = wid / heads;
  int h = (int)(wid % heads);
  int hd = d / heads;
  bool on = lane < hd;
  float scale = rsqrtf((float)d);
  float qv[3], kv[3], vv[3];
  for (int c = 0; c < C; ++c) {
    long base = (node * C + c) * d + h * hd + lane;
    qv[c] = on ? q[base] : 0.f;
    kv[c] = on ? k[base] : 0.f;
    vv[c] = on ? v[base] : 0.f;
  }
  for (int qi = 0; qi < C; ++qi) {
    float sc[3];
    for (int ki = 0; ki < C; ++ki) {
      float p = qv[qi] * kv[ki];
      for (int off = 16; off > 0; off >>= 1) p += __shfl_down(p, off, 32);
      sc[ki] = __shfl(p, 0, 32) * scale;
    }
    float mx = sc[0];
    for (int ki = 1; ki < C; ++ki) mx = fmaxf(mx, sc[ki]);
    float e[3], tot = 0.f, o = 0.f;
    for (int ki = 0; ki < C; ++ki) { e[ki] = __expf(sc[ki] - mx); tot += e[ki]; }
    for (int ki = 0; ki < C; ++ki) o += e[ki] * vv[ki];
    o /= tot;
    if (on) outb[(node * C + qi) * d + h * hd + lane] = f2bf(o);
  }
}

// pooling attention: only query row 0; also accumulates head-mean probs a0[N,C]
__global__ void k_pool_attn(const float* __restrict__ q, const float* __restrict__ k,
                            const float* __restrict__ v, u16* __restrict__ out0,
                            float* __restrict__ a0, long N, int C, int d, int heads)
{
  long gt = (long)blockIdx.x * blockDim.x + threadIdx.x;
  long wid = gt >> 5;
  int lane = threadIdx.x & 31;
  if (wid >= N * heads) return;
  long node = wid / heads;
  int h = (int)(wid % heads);
  int hd = d / heads;
  bool on = lane < hd;
  float scale = rsqrtf((float)d);
  long b0 = (node * C) * d + h * hd + lane;
  float qv0 = on ? q[b0] : 0.f;
  float kv[3], vv[3];
  for (int c = 0; c < C; ++c) {
    long base = (node * C + c) * d + h * hd + lane;
    kv[c] = on ? k[base] : 0.f;
    vv[c] = on ? v[base] : 0.f;
  }
  float sc[3];
  for (int ki = 0; ki < C; ++ki) {
    float p = qv0 * kv[ki];
    for (int off = 16; off > 0; off >>= 1) p += __shfl_down(p, off, 32);
    sc[ki] = __shfl(p, 0, 32) * scale;
  }
  float mx = sc[0];
  for (int ki = 1; ki < C; ++ki) mx = fmaxf(mx, sc[ki]);
  float e[3], tot = 0.f, o = 0.f;
  for (int ki = 0; ki < C; ++ki) { e[ki] = __expf(sc[ki] - mx); tot += e[ki]; }
  for (int ki = 0; ki < C; ++ki) o += (e[ki] / tot) * vv[ki];
  if (on) out0[node * d + h * hd + lane] = f2bf(o);
  if (lane == 0)
    for (int ki = 0; ki < C; ++ki)
      atomicAdd(&a0[node * C + ki], (e[ki] / tot) / heads);
}

// v0[m,:] = sum_c a0[m,c] * v[m,c,:]
__global__ void k_poolv(const float* __restrict__ a0, const float* __restrict__ v,
                        float* __restrict__ out, long N, int C, int W)
{
  long i = (long)blockIdx.x * blockDim.x + threadIdx.x;
  if (i >= N * W) return;
  long m = i / W; int j = (int)(i % W);
  float acc = 0.f;
  for (int c = 0; c < C; ++c) acc += a0[m * C + c] * v[(m * C + c) * W + j];
  out[i] = acc;
}

// pes_cat = [pes(32) | fwd ? Ws[seq[src]] : 0 (4)] as bf16, ld with zero pad
__global__ void k_pes(const float* __restrict__ pes, const float* __restrict__ Ws,
                      const int* __restrict__ seq, const int* __restrict__ src,
                      const int* __restrict__ dst, u16* __restrict__ out,
                      long E, int ld)
{
  int W = SE + ODIM;
  long i = (long)blockIdx.x * blockDim.x + threadIdx.x;
  if (i >= E * ld) return;
  long e = i / ld; int c = (int)(i % ld);
  if (c >= W) { out[i] = 0; return; }
  float val;
  if (c < SE) val = pes[e * SE + c];
  else {
    bool fwd = src[e] < dst[e];
    val = fwd ? Ws[(long)seq[src[e]] * ODIM + (c - SE)] : 0.f;
  }
  out[i] = f2bf(val);
}

// ---------------------------------------------------------------------------
// Host-side orchestration
// ---------------------------------------------------------------------------
struct Arena { char* p; };
static inline void* aal(Arena& a, size_t bytes) {
  void* r = a.p; a.p += (bytes + 255) & ~(size_t)255; return r;
}
static inline unsigned gb(long n, int b) { return (unsigned)((n + b - 1) / b); }
static inline int rup(int x, int a) { return (x + a - 1) / a * a; }

static void gemm(hipStream_t st, const u16* A, const u16* B, const float* bias,
                 float* Cf, u16* Cb, int ldcb, long M, int N, int K, int act,
                 const float* Res)
{
  int kp = rup(K, 32);   // caller guarantees A's leading dim == kp
  dim3 g((unsigned)((N + TN - 1) / TN), (unsigned)((M + TM - 1) / TM));
  k_gemm<<<g, 256, 0, st>>>(A, B, bias, Cf, Cb, ldcb, (int)M, N, kp, act, Res);
}

struct Cur { void* const* d; int i; const float* f() { return (const float*)d[i++]; } };
struct LNW  { const float* g; const float* b; };
struct GVPW { const u16* wh; const u16* wsw; const float* wsb;
              const u16* wsvw; const float* wsvb; const u16* wv;
              int si, vi, h, so, vo; };
struct MHAW { const u16* q; const u16* k; const u16* v; const u16* o; const float* ob; };
struct EncL { MHAW attn; LNW an; GVPW conv[3]; GVPW ff0, ff1; LNW n0, n1; };
struct DecL { GVPW conv[3]; GVPW ff0, ff1; LNW n0, n1; };

// pack a (K,N) f32 weight into WMMA-friendly bf16 [rup64(N)][rup32(K)]
static const u16* wpack(hipStream_t st, Arena& a, const float* w, int K, int N) {
  int kp = rup(K, 32), np = rup(N, 64);
  long tot = (long)np * kp;
  u16* p = (u16*)aal(a, (size_t)tot * 2);
  k_pack<<<gb(tot, 256), 256, 0, st>>>(w, p, K, N, kp, tot);
  return p;
}
static void read_ln(Cur& c, LNW& o) { o.b = c.f(); o.g = c.f(); }  // sorted: b,g
static void read_gvp(Cur& c, hipStream_t st, Arena& a, GVPW& g,
                     int si, int vi, int so, int vo, bool gate, bool haswv)
{ // sorted keys: wh, ws{b,w}, wsv{b,w}, wv
  g.si = si; g.vi = vi; g.so = so; g.vo = vo;
  g.h = (vi > vo) ? vi : vo;
  g.wh  = wpack(st, a, c.f(), vi, g.h);
  g.wsb = c.f();
  g.wsw = wpack(st, a, c.f(), si + g.h, so);
  if (gate) { g.wsvb = c.f(); g.wsvw = wpack(st, a, c.f(), so, vo); }
  else      { g.wsvb = nullptr; g.wsvw = nullptr; }
  g.wv = haswv ? wpack(st, a, c.f(), g.h, vo) : nullptr;
}
static void read_mha(Cur& c, hipStream_t st, Arena& a, MHAW& m, int d)
{ // sorted keys: k, o{b,w}, q, v
  m.k  = wpack(st, a, c.f(), d, d);
  m.ob = c.f();
  m.o  = wpack(st, a, c.f(), d, d);
  m.q  = wpack(st, a, c.f(), d, d);
  m.v  = wpack(st, a, c.f(), d, d);
}
static void read_enc(Cur& c, hipStream_t st, Arena& a, EncL& L)
{ // sorted: attn, attn_norm, conv, ff0, ff1, norm0, norm1
  read_mha(c, st, a, L.attn, NS);
  read_ln(c, L.an);
  read_gvp(c, st, a, L.conv[0], 2 * NS + SE, 2 * NV + VE, NS, NV, true, true);
  read_gvp(c, st, a, L.conv[1], NS, NV, NS, NV, true, true);
  read_gvp(c, st, a, L.conv[2], NS, NV, NS, NV, false, true);
  read_gvp(c, st, a, L.ff0, NS, NV, 4 * NS, 2 * NV, true, true);
  read_gvp(c, st, a, L.ff1, 4 * NS, 2 * NV, NS, NV, false, true);
  read_ln(c, L.n0);
  read_ln(c, L.n1);
}
static void read_dec(Cur& c, hipStream_t st, Arena& a, DecL& L)
{ // sorted: conv, ff0, ff1, norm0, norm1
  read_gvp(c, st, a, L.conv[0], 2 * NS + SE + ODIM, 2 * NV + VE, NS, NV, true, true);
  read_gvp(c, st, a, L.conv[1], NS, NV, NS, NV, true, true);
  read_gvp(c, st, a, L.conv[2], NS, NV, NS, NV, false, true);
  read_gvp(c, st, a, L.ff0, NS, NV, 4 * NS, 2 * NV, true, true);
  read_gvp(c, st, a, L.ff1, 4 * NS, 2 * NV, NS, NV, false, true);
  read_ln(c, L.n0);
  read_ln(c, L.n1);
}

// Full GVP: s_out = act(ws @ [s, ||v^T wh||] + b)  (+s_res),
//           v_out = (v^T wh wv)^T * sigmoid(wsv @ s_pre)  (+v_res)
static void run_gvp(hipStream_t st, Arena sc, const GVPW& g, long rows,
                    const u16* s_bf, int s_ld, const float* v_in, int act,
                    float* s_out_f, u16* s_out_b, float* v_out,
                    const float* s_res, const float* v_res)
{
  long r3 = rows * 3;
  const u16* cat = s_bf; int catw = g.si;
  u16* vhb = nullptr;
  int hp = rup(g.h, 32);
  if (g.vi > 0) {
    int vp = rup(g.vi, 32);
    u16* vT = (u16*)aal(sc, (size_t)r3 * vp * 2);
    k_vT<<<gb(r3 * vp, 256), 256, 0, st>>>(v_in, vT, rows, g.vi, vp);
    float* vh = (float*)aal(sc, (size_t)r3 * g.h * 4);
    vhb = g.wv ? (u16*)aal(sc, (size_t)r3 * hp * 2) : nullptr;
    gemm(st, vT, g.wh, nullptr, vh, vhb, hp, r3, g.h, g.vi, 0, nullptr);
    catw = g.si + g.h;
    int catp = rup(catw, 32);
    u16* cc = (u16*)aal(sc, (size_t)rows * catp * 2);
    k_copyc<<<gb(rows * g.si, 256), 256, 0, st>>>(s_bf, s_ld, cc, catp, rows, g.si, 0);
    k_vhnorm<<<gb(rows * g.h, 256), 256, 0, st>>>(vh, cc, catp, g.si, rows, g.h);
    if (catp > catw)
      k_padz<<<gb(rows * (catp - catw), 256), 256, 0, st>>>(cc, rows, catw, catp);
    cat = cc;
  }
  if (g.wsvw) { // vector-gated path: keep pre-activation s for the gate
    float* spre = (float*)aal(sc, (size_t)rows * g.so * 4);
    u16*  spreb = (u16*)aal(sc, (size_t)rows * g.so * 2);
    gemm(st, cat, g.wsw, g.wsb, spre, spreb, g.so, rows, g.so, catw, 0, nullptr);
    float* gate = (float*)aal(sc, (size_t)rows * g.vo * 4);
    gemm(st, spreb, g.wsvw, g.wsvb, gate, nullptr, 0, rows, g.vo, g.so, 2, nullptr);
    k_act<<<gb(rows * (long)g.so, 256), 256, 0, st>>>(spre, s_res, s_out_f, s_out_b,
                                                      rows * (long)g.so, act);
    if (g.wv && v_out) {
      float* vot = (float*)aal(sc, (size_t)r3 * g.vo * 4);
      gemm(st, vhb, g.wv, nullptr, vot, nullptr, 0, r3, g.vo, g.h, 0, nullptr);
      k_gatev<<<gb(rows * g.vo * 3, 256), 256, 0, st>>>(vot, gate, v_res, v_out, rows, g.vo);
    }
  } else {
    gemm(st, cat, g.wsw, g.wsb, s_out_f, s_out_b, rup(g.so, 32), rows, g.so, catw,
         act, s_res);
    if (g.wv && v_out) {
      float* vot = (float*)aal(sc, (size_t)r3 * g.vo * 4);
      gemm(st, vhb, g.wv, nullptr, vot, nullptr, 0, r3, g.vo, g.h, 0, nullptr);
      k_gatev<<<gb(rows * g.vo * 3, 256), 256, 0, st>>>(vot, nullptr, v_res, v_out, rows, g.vo);
    }
  }
}

static void enc_layer(hipStream_t st, Arena sc, const EncL& L,
                      float* s, float* v, const u16* es_bf, const float* ev,
                      const int* src, const int* dst, const float* cnt)
{
  long R = (long)NN * NC, Rm = (long)NE * NC;
  u16* sn = (u16*)aal(sc, (size_t)R * NS * 2);
  k_ln<<<gb(R, 8), 256, 0, st>>>(s, L.n0.g, L.n0.b, nullptr, sn, R, NS);
  float* vn = (float*)aal(sc, (size_t)R * NV * 3 * 4);
  k_vnormv<<<gb(R, 256), 256, 0, st>>>(v, vn, R, NV);

  int W = 2 * NS + SE;      // 288, already a multiple of 32
  u16* ms = (u16*)aal(sc, (size_t)Rm * W * 2);
  k_msgs<<<gb(Rm * W, 256), 256, 0, st>>>(sn, nullptr, es_bf, src, dst, ms,
                                          NE, NC, NS, SE, SE, W);
  float* mv = (float*)aal(sc, (size_t)Rm * 33 * 3 * 4);
  k_msgv<<<gb(Rm * 33 * 3, 256), 256, 0, st>>>(vn, nullptr, ev, src, dst, mv, NE, NC, NV, VE);

  float* a_s = (float*)aal(sc, (size_t)Rm * NS * 4);
  u16*  a_sb = (u16*)aal(sc, (size_t)Rm * NS * 2);
  float* a_v = (float*)aal(sc, (size_t)Rm * NV * 3 * 4);
  run_gvp(st, sc, L.conv[0], Rm, ms, W, mv, 1, a_s, a_sb, a_v, nullptr, nullptr);
  float* b_s = (float*)aal(sc, (size_t)Rm * NS * 4);
  u16*  b_sb = (u16*)aal(sc, (size_t)Rm * NS * 2);
  float* b_v = (float*)aal(sc, (size_t)Rm * NV * 3 * 4);
  run_gvp(st, sc, L.conv[1], Rm, a_sb, NS, a_v, 1, b_s, b_sb, b_v, nullptr, nullptr);
  run_gvp(st, sc, L.conv[2], Rm, b_sb, NS, b_v, 0, a_s, nullptr, a_v, nullptr, nullptr);

  float* aggs = (float*)aal(sc, (size_t)R * NS * 4);
  float* aggv = (float*)aal(sc, (size_t)R * NV * 3 * 4);
  hipMemsetAsync(aggs, 0, (size_t)R * NS * 4, st);
  hipMemsetAsync(aggv, 0, (size_t)R * NV * 3 * 4, st);
  k_scat<<<gb(Rm * NS, 256), 256, 0, st>>>(a_s, dst, aggs, NE, NC, NS);
  k_scat<<<gb(Rm * NV * 3, 256), 256, 0, st>>>(a_v, dst, aggv, NE, NC, NV * 3);
  k_addsc<<<gb(R * NS, 256), 256, 0, st>>>(aggs, cnt, s, R, NC, NS);
  k_addsc<<<gb(R * NV * 3, 256), 256, 0, st>>>(aggv, cnt, v, R, NC, NV * 3);

  // conformer self-attention + residual
  u16* xb = (u16*)aal(sc, (size_t)R * NS * 2);
  k_ln<<<gb(R, 8), 256, 0, st>>>(s, L.an.g, L.an.b, nullptr, xb, R, NS);
  float* q  = (float*)aal(sc, (size_t)R * NS * 4);
  float* kk = (float*)aal(sc, (size_t)R * NS * 4);
  float* vv = (float*)aal(sc, (size_t)R * NS * 4);
  gemm(st, xb, L.attn.q, nullptr, q,  nullptr, 0, R, NS, NS, 0, nullptr);
  gemm(st, xb, L.attn.k, nullptr, kk, nullptr, 0, R, NS, NS, 0, nullptr);
  gemm(st, xb, L.attn.v, nullptr, vv, nullptr, 0, R, NS, NS, 0, nullptr);
  u16* ao = (u16*)aal(sc, (size_t)R * NS * 2);
  k_attn<<<gb((long)NN * NHEADS * 32, 256), 256, 0, st>>>(q, kk, vv, ao, NN, NC, NS, NHEADS);
  gemm(st, ao, L.attn.o, L.attn.ob, s, nullptr, 0, R, NS, NS, 0, s);

  // feed-forward GVPs + residual
  u16* sn2 = (u16*)aal(sc, (size_t)R * NS * 2);
  k_ln<<<gb(R, 8), 256, 0, st>>>(s, L.n1.g, L.n1.b, nullptr, sn2, R, NS);
  float* vn2 = (float*)aal(sc, (size_t)R * NV * 3 * 4);
  k_vnormv<<<gb(R, 256), 256, 0, st>>>(v, vn2, R, NV);
  float* fs  = (float*)aal(sc, (size_t)R * 4 * NS * 4);
  u16*  fsb  = (u16*)aal(sc, (size_t)R * 4 * NS * 2);
  float* fv  = (float*)aal(sc, (size_t)R * 2 * NV * 3 * 4);
  run_gvp(st, sc, L.ff0, R, sn2, NS, vn2, 1, fs, fsb, fv, nullptr, nullptr);
  run_gvp(st, sc, L.ff1, R, fsb, 4 * NS, fv, 0, s, nullptr, v, s, v);
}

static void dec_layer(hipStream_t st, Arena sc, const DecL& L,
                      float* s, float* v, const float* ars, const float* arv,
                      const u16* pes, const float* pev,
                      const int* src, const int* dst, const float* cnt)
{
  long R = NN, Rm = NE;
  u16* sn = (u16*)aal(sc, (size_t)R * NS * 2);
  k_ln<<<gb(R, 8), 256, 0, st>>>(s, L.n0.g, L.n0.b, nullptr, sn, R, NS);
  u16* an = (u16*)aal(sc, (size_t)R * NS * 2);
  k_ln<<<gb(R, 8), 256, 0, st>>>(ars, L.n0.g, L.n0.b, nullptr, an, R, NS);
  float* vn = (float*)aal(sc, (size_t)R * NV * 3 * 4);
  k_vnormv<<<gb(R, 256), 256, 0, st>>>(v, vn, R, NV);
  float* av = (float*)aal(sc, (size_t)R * NV * 3 * 4);
  k_vnormv<<<gb(R, 256), 256, 0, st>>>(arv, av, R, NV);

  int ESD = SE + ODIM;           // 36
  int W = 2 * NS + ESD;          // 292
  int Wp = rup(W, 32);           // 320
  u16* ms = (u16*)aal(sc, (size_t)Rm * Wp * 2);
  k_msgs<<<gb(Rm * Wp, 256), 256, 0, st>>>(sn, an, pes, src, dst, ms,
                                           NE, 1, NS, ESD, rup(ESD, 32), Wp);
  float* mv = (float*)aal(sc, (size_t)Rm * 33 * 3 * 4);
  k_msgv<<<gb(Rm * 33 * 3, 256), 256, 0, st>>>(vn, av, pev, src, dst, mv, NE, 1, NV, VE);

  float* a_s = (float*)aal(sc, (size_t)Rm * NS * 4);
  u16*  a_sb = (u16*)aal(sc, (size_t)Rm * NS * 2);
  float* a_v = (float*)aal(sc, (size_t)Rm * NV * 3 * 4);
  run_gvp(st, sc, L.conv[0], Rm, ms, Wp, mv, 1, a_s, a_sb, a_v, nullptr, nullptr);
  float* b_s = (float*)aal(sc, (size_t)Rm * NS * 4);
  u16*  b_sb = (u16*)aal(sc, (size_t)Rm * NS * 2);
  float* b_v = (float*)aal(sc, (size_t)Rm * NV * 3 * 4);
  run_gvp(st, sc, L.conv[1], Rm, a_sb, NS, a_v, 1, b_s, b_sb, b_v, nullptr, nullptr);
  run_gvp(st, sc, L.conv[2], Rm, b_sb, NS, b_v, 0, a_s, nullptr, a_v, nullptr, nullptr);

  float* aggs = (float*)aal(sc, (size_t)R * NS * 4);
  float* aggv = (float*)aal(sc, (size_t)R * NV * 3 * 4);
  hipMemsetAsync(aggs, 0, (size_t)R * NS * 4, st);
  hipMemsetAsync(aggv, 0, (size_t)R * NV * 3 * 4, st);
  k_scat<<<gb(Rm * NS, 256), 256, 0, st>>>(a_s, dst, aggs, NE, 1, NS);
  k_scat<<<gb(Rm * NV * 3, 256), 256, 0, st>>>(a_v, dst, aggv, NE, 1, NV * 3);
  k_addsc<<<gb(R * NS, 256), 256, 0, st>>>(aggs, cnt, s, R, 1, NS);
  k_addsc<<<gb(R * NV * 3, 256), 256, 0, st>>>(aggv, cnt, v, R, 1, NV * 3);

  u16* sn2 = (u16*)aal(sc, (size_t)R * NS * 2);
  k_ln<<<gb(R, 8), 256, 0, st>>>(s, L.n1.g, L.n1.b, nullptr, sn2, R, NS);
  float* vn2 = (float*)aal(sc, (size_t)R * NV * 3 * 4);
  k_vnormv<<<gb(R, 256), 256, 0, st>>>(v, vn2, R, NV);
  float* fs = (float*)aal(sc, (size_t)R * 4 * NS * 4);
  u16*  fsb = (u16*)aal(sc, (size_t)R * 4 * NS * 2);
  float* fv = (float*)aal(sc, (size_t)R * 2 * NV * 3 * 4);
  run_gvp(st, sc, L.ff0, R, sn2, NS, vn2, 1, fs, fsb, fv, nullptr, nullptr);
  run_gvp(st, sc, L.ff1, R, fsb, 4 * NS, fv, 0, s, nullptr, v, s, v);
}

extern "C" void kernel_launch(void* const* d_in, const int* in_sizes, int n_in,
                              void* d_out, int out_size, void* d_ws, size_t ws_size,
                              hipStream_t stream)
{
  (void)in_sizes; (void)n_in; (void)out_size; (void)ws_size;
  Arena per { (char*)d_ws };
  const float* node_s = (const float*)d_in[0];
  const float* node_v = (const float*)d_in[1];
  const float* edge_s = (const float*)d_in[2];
  const float* edge_v = (const float*)d_in[3];

  // ---- parameters (pytree leaves, sorted-dict order), weights -> packed bf16
  Cur c { d_in, 4 };
  GVPW Weg; LNW Weln; GVPW Woutg; const float* Ws; GVPW Wvg; LNW Wvln;
  DecL dec[3]; MHAW epool; EncL enc[3]; MHAW npool;
  read_gvp(c, stream, per, Weg, SE, VE, SE, VE, true, true);      // We.gvp
  read_ln(c, Weln);                                               // We.ln
  read_gvp(c, stream, per, Woutg, NS, NV, ODIM, 0, false, false); // Wout
  Ws = c.f();                                                     // Ws (4x4)
  read_gvp(c, stream, per, Wvg, 64, 4, NS, NV, true, true);       // Wv.gvp
  read_ln(c, Wvln);                                               // Wv.ln
  for (int l = 0; l < 3; ++l) read_dec(c, stream, per, dec[l]);
  read_mha(c, stream, per, epool, SE);
  for (int l = 0; l < 3; ++l) read_enc(c, stream, per, enc[l]);
  read_mha(c, stream, per, npool, NS);
  const int* eidx = (const int*)d_in[c.i + 0];
  const int* seq  = (const int*)d_in[c.i + 1];
  const int* src = eidx, *dst = eidx + NE;
  // mask_confs is all-true in the reference setup; pooling bias == 0.

  long R = (long)NN * NC, Rm = (long)NE * NC;
  // ---- persistent state buffers
  float* s   = (float*)aal(per, (size_t)R * NS * 4);
  float* v   = (float*)aal(per, (size_t)R * NV * 3 * 4);
  u16*  esb  = (u16*)aal(per, (size_t)Rm * SE * 2);
  float* ev  = (float*)aal(per, (size_t)Rm * VE * 3 * 4);
  float* cnt = (float*)aal(per, (size_t)NN * 4);
  float* ps  = (float*)aal(per, (size_t)NN * NS * 4);
  float* pv  = (float*)aal(per, (size_t)NN * NV * 3 * 4);
  u16*  pesc = (u16*)aal(per, (size_t)NE * rup(SE + ODIM, 32) * 2);
  float* pev = (float*)aal(per, (size_t)NE * VE * 3 * 4);
  float* hs  = (float*)aal(per, (size_t)NN * NS * 4);
  float* hv  = (float*)aal(per, (size_t)NN * NV * 3 * 4);
  Arena scr = per;  // everything past here is reusable scratch (in-order stream)

  hipMemsetAsync(cnt, 0, (size_t)NN * 4, stream);
  k_count<<<gb(NE, 256), 256, 0, stream>>>(dst, cnt, NE);

  { // node embedding: gvp_norm(Wv.ln) + GVP
    Arena sc = scr;
    u16* nln = (u16*)aal(sc, (size_t)R * 64 * 2);
    k_ln<<<gb(R, 8), 256, 0, stream>>>(node_s, Wvln.g, Wvln.b, nullptr, nln, R, 64);
    float* nvn = (float*)aal(sc, (size_t)R * 4 * 3 * 4);
    k_vnormv<<<gb(R, 256), 256, 0, stream>>>(node_v, nvn, R, 4);
    run_gvp(stream, sc, Wvg, R, nln, 64, nvn, 0, s, nullptr, v, nullptr, nullptr);
  }
  { // edge embedding
    Arena sc = scr;
    u16* eln = (u16*)aal(sc, (size_t)Rm * SE * 2);
    k_ln<<<gb(Rm, 8), 256, 0, stream>>>(edge_s, Weln.g, Weln.b, nullptr, eln, Rm, SE);
    float* evn = (float*)aal(sc, (size_t)Rm * VE * 3 * 4);
    k_vnormv<<<gb(Rm, 256), 256, 0, stream>>>(edge_v, evn, Rm, VE);
    run_gvp(stream, sc, Weg, Rm, eln, SE, evn, 0, nullptr, esb, ev, nullptr, nullptr);
  }

  for (int l = 0; l < 3; ++l)
    enc_layer(stream, scr, enc[l], s, v, esb, ev, src, dst, cnt);

  { // node + edge pooling over conformers
    Arena sc = scr;
    u16* sb = (u16*)aal(sc, (size_t)R * NS * 2);
    k_cvt<<<gb(R * NS, 256), 256, 0, stream>>>(s, sb, R * NS);
    float* q  = (float*)aal(sc, (size_t)R * NS * 4);
    float* kk = (float*)aal(sc, (size_t)R * NS * 4);
    float* vv = (float*)aal(sc, (size_t)R * NS * 4);
    gemm(stream, sb, npool.q, nullptr, q,  nullptr, 0, R, NS, NS, 0, nullptr);
    gemm(stream, sb, npool.k, nullptr, kk, nullptr, 0, R, NS, NS, 0, nullptr);
    gemm(stream, sb, npool.v, nullptr, vv, nullptr, 0, R, NS, NS, 0, nullptr);
    float* a0 = (float*)aal(sc, (size_t)NN * NC * 4);
    hipMemsetAsync(a0, 0, (size_t)NN * NC * 4, stream);
    u16* o0 = (u16*)aal(sc, (size_t)NN * NS * 2);
    k_pool_attn<<<gb((long)NN * NHEADS * 32, 256), 256, 0, stream>>>(q, kk, vv, o0, a0,
                                                                     NN, NC, NS, NHEADS);
    gemm(stream, o0, npool.o, npool.ob, ps, nullptr, 0, NN, NS, NS, 0, nullptr);
    k_poolv<<<gb((long)NN * NV * 3, 256), 256, 0, stream>>>(a0, v, pv, NN, NC, NV * 3);

    float* eq = (float*)aal(sc, (size_t)Rm * SE * 4);
    float* ek = (float*)aal(sc, (size_t)Rm * SE * 4);
    float* ew = (float*)aal(sc, (size_t)Rm * SE * 4);
    gemm(stream, esb, epool.q, nullptr, eq, nullptr, 0, Rm, SE, SE, 0, nullptr);
    gemm(stream, esb, epool.k, nullptr, ek, nullptr, 0, Rm, SE, SE, 0, nullptr);
    gemm(stream, esb, epool.v, nullptr, ew, nullptr, 0, Rm, SE, SE, 0, nullptr);
    float* ea0 = (float*)aal(sc, (size_t)NE * NC * 4);
    hipMemsetAsync(ea0, 0, (size_t)NE * NC * 4, stream);
    u16* eo0 = (u16*)aal(sc, (size_t)NE * SE * 2);
    k_pool_attn<<<gb((long)NE * NHEADS * 32, 256), 256, 0, stream>>>(eq, ek, ew, eo0, ea0,
                                                                     NE, NC, SE, NHEADS);
    float* pes = (float*)aal(sc, (size_t)NE * SE * 4);
    gemm(stream, eo0, epool.o, epool.ob, pes, nullptr, 0, NE, SE, SE, 0, nullptr);
    k_poolv<<<gb((long)NE * VE * 3, 256), 256, 0, stream>>>(ea0, ev, pev, NE, NC, VE * 3);
    k_pes<<<gb((long)NE * rup(SE + ODIM, 32), 256), 256, 0, stream>>>(
        pes, Ws, seq, src, dst, pesc, NE, rup(SE + ODIM, 32));
  }

  hipMemcpyAsync(hs, ps, (size_t)NN * NS * 4, hipMemcpyDeviceToDevice, stream);
  hipMemcpyAsync(hv, pv, (size_t)NN * NV * 3 * 4, hipMemcpyDeviceToDevice, stream);
  for (int l = 0; l < 3; ++l)
    dec_layer(stream, scr, dec[l], hs, hv, ps, pv, pesc, pev, src, dst, cnt);

  { // Wout GVP -> logits (NN x 4) straight into d_out
    Arena sc = scr;
    u16* hb = (u16*)aal(sc, (size_t)NN * NS * 2);
    k_cvt<<<gb((long)NN * NS, 256), 256, 0, stream>>>(hs, hb, (long)NN * NS);
    run_gvp(stream, sc, Woutg, NN, hb, NS, hv, 0, (float*)d_out, nullptr, nullptr,
            nullptr, nullptr);
  }
}
